// Seq2SeqLSTM_41257455846027
// MI455X (gfx1250) — matrix operand order repaired and checked by reading, not compiled, not run on hardware
//
#include <hip/hip_runtime.h>
#include <hip/hip_bf16.h>
#include <math.h>

// Problem constants (from reference)
#define VV 32000
#define EE 128
#define HH 256
#define BB 32
#define SS 64
#define TT 64
#define G4H 1024   // 4*H

typedef __attribute__((ext_vector_type(16))) _Float16 v16h;
typedef __attribute__((ext_vector_type(8)))  _Float16 v8h;
typedef __attribute__((ext_vector_type(8)))  float    v8f;

// ---------------------------------------------------------------------------
// Tile loaders per CDNA5 WMMA VGPR layouts (CDNA5 ISA §7.12.2)
// A: 16x32 f16, lane<16 holds row (lane&15), K = {k..k+7, k+16..k+23};
//    lane>=16 same row, K = {k+8..k+15, k+24..k+31}.
// B: 32x16 f16 as columns: lane&15 = column n (row n of W[N,K]),
//    same K split as A: lane<16 -> K = {k..k+7, k+16..k+23}, lane>=16 -> {k+8..k+15, k+24..k+31}.
// ---------------------------------------------------------------------------
__device__ __forceinline__ v16h load_tile_a(const _Float16* base, int ld, int lane, int k) {
    const int row = lane & 15;
    const int sel = lane >> 4;
    const _Float16* p = base + (size_t)row * ld + k + sel * 8;
    union { v16h v; v8h h[2]; } u;
    u.h[0] = *(const v8h*)(p);
    u.h[1] = *(const v8h*)(p + 16);
    return u.v;
}

__device__ __forceinline__ v16h load_tile_b(const _Float16* base, int ld, int lane, int k) {
    const int col = lane & 15;
    const int sel = lane >> 4;
    const _Float16* p = base + (size_t)col * ld + k + sel * 8;
    union { v16h v; v8h h[2]; } u;
    u.h[0] = *(const v8h*)(p);
    u.h[1] = *(const v8h*)(p + 16);
    return u.v;
}

__device__ __forceinline__ float sigm(float x) { return 1.0f / (1.0f + __expf(-x)); }

// ---------------------------------------------------------------------------
// f32 -> f16 elementwise convert (weights)
// ---------------------------------------------------------------------------
__global__ void k_f32_to_f16(const float* __restrict__ in, _Float16* __restrict__ out, int n) {
    int i = blockIdx.x * blockDim.x + threadIdx.x;
    if (i < n) out[i] = (_Float16)in[i];
}

// ---------------------------------------------------------------------------
// Embedding gather + convert: out[row=t*32+b][e] = emb[tok[b*seqStride+t]][e]
// grid = rows, block = 128 (=E)
// ---------------------------------------------------------------------------
__global__ void k_gather_embed(const float* __restrict__ emb, const int* __restrict__ tok,
                               _Float16* __restrict__ out, int seqStride) {
    const int row = blockIdx.x;
    const int t = row >> 5;
    const int b = row & 31;
    const int id = tok[b * seqStride + t];
    out[(size_t)row * EE + threadIdx.x] = (_Float16)emb[(size_t)id * EE + threadIdx.x];
}

// ---------------------------------------------------------------------------
// Input GEMM: C[M,1024] = A[M,128] * W[1024,128]^T (zero init), K=128 fixed.
// Each wave: 2 m-tiles x 4 n-tiles -> 8 WMMAs per k-step, 6 tile-loads
// (B reused across the two m-tiles, A across the four n-tiles).
// ---------------------------------------------------------------------------
__global__ __launch_bounds__(256) void k_gemm_xw(const _Float16* __restrict__ A,
                                                 const _Float16* __restrict__ W,
                                                 float* __restrict__ C,
                                                 int Mpairs, int Ngroups) {
    const int wave = threadIdx.x >> 5;
    const int lane = threadIdx.x & 31;
    const int job  = blockIdx.x * 8 + wave;
    if (job >= Mpairs * Ngroups) return;               // wave-uniform exit
    const int m0 = (job / Ngroups) * 32;               // 2 m-tiles of 16
    const int n0 = (job % Ngroups) * 64;               // 4 n-tiles of 16
    v8f acc[8] = {};                                   // [j] = m-tile0, [4+j] = m-tile1
#pragma unroll
    for (int k = 0; k < EE; k += 32) {
        v16h a0 = load_tile_a(A + (size_t)m0 * EE,        EE, lane, k);
        v16h a1 = load_tile_a(A + (size_t)(m0 + 16) * EE, EE, lane, k);
#pragma unroll
        for (int j = 0; j < 4; ++j) {
            v16h bv = load_tile_b(W + (size_t)(n0 + 16 * j) * EE, EE, lane, k);
            acc[j]     = __builtin_amdgcn_wmma_f32_16x16x32_f16(false, a0, false, bv,
                                                                (short)0, acc[j], false, false);
            acc[4 + j] = __builtin_amdgcn_wmma_f32_16x16x32_f16(false, a1, false, bv,
                                                                (short)0, acc[4 + j], false, false);
        }
    }
    const int mb = (lane >> 4) * 8;
#pragma unroll
    for (int j = 0; j < 8; ++j) {
        const int n  = n0 + 16 * (j & 3) + (lane & 15);
        const int mm = m0 + 16 * (j >> 2) + mb;
#pragma unroll
        for (int r = 0; r < 8; ++r)
            C[(size_t)(mm + r) * G4H + n] = acc[j][r];
    }
}

// ---------------------------------------------------------------------------
// LSTM recurrence: one workgroup (1024 thr = 32 waves = full WGP, 8/SIMD).
// Gate-major wave assignment: wave (mT, jt) owns ALL FOUR gate tiles of its
// (m, j) patch -> i/f/g/o land in that wave's registers, cell update is
// register-resident. h (f16) + c (f32) live in LDS; no global gate bounce.
// ---------------------------------------------------------------------------
#define HJ_LD 264   // 256 + 8 halfs pad -> 16 A-rows hit disjoint bank groups

__global__ __launch_bounds__(1024) void k_lstm(const float* __restrict__ xg,        // [steps,32,1024]
                                               const _Float16* __restrict__ Whh16,  // [1024,256]
                                               const float* __restrict__ bias,      // [1024]
                                               const float* __restrict__ h_in,      // [32,256] or null
                                               const float* __restrict__ c_in,      // [32,256] or null
                                               float* __restrict__ h_out,           // [32,256] or null
                                               float* __restrict__ c_out,           // [32,256] or null
                                               _Float16* __restrict__ hs16,         // [steps,32,256] or null
                                               int steps) {
    __shared__ _Float16 hj[BB * HJ_LD];
    __shared__ float    cst[BB * HH];
    const int tid  = threadIdx.x;
    const int lane = tid & 31;
    const int wave = tid >> 5;
    const int m0   = (wave >> 4) * 16;        // waves 0-15 -> rows 0..15, 16-31 -> 16..31
    const int j0   = (wave & 15) * 16;        // j-slice owned by this wave
    const int jcol = j0 + (lane & 15);
    const int mb   = (lane >> 4) * 8;

    for (int idx = tid; idx < BB * HH; idx += 1024) {
        const int b = idx >> 8, j = idx & 255;
        hj[b * HJ_LD + j] = (_Float16)(h_in ? h_in[idx] : 0.0f);
        cst[idx]          = c_in ? c_in[idx] : 0.0f;
    }
    __syncthreads();

    const float bi = bias[jcol];
    const float bf = bias[HH + jcol];
    const float bg = bias[2 * HH + jcol];
    const float bo = bias[3 * HH + jcol];
    const _Float16* Wi = Whh16 + (size_t)(0 * HH + j0) * HH;
    const _Float16* Wf = Whh16 + (size_t)(1 * HH + j0) * HH;
    const _Float16* Wg = Whh16 + (size_t)(2 * HH + j0) * HH;
    const _Float16* Wo = Whh16 + (size_t)(3 * HH + j0) * HH;

    for (int t = 0; t < steps; ++t) {
        const float* xgt = xg + (size_t)t * BB * G4H;
        v8f ai, af, ag, ao;
#pragma unroll
        for (int r = 0; r < 8; ++r) {
            const float* xr = xgt + (size_t)(m0 + mb + r) * G4H;
            ai[r] = xr[jcol]          + bi;
            af[r] = xr[HH + jcol]     + bf;
            ag[r] = xr[2 * HH + jcol] + bg;
            ao[r] = xr[3 * HH + jcol] + bo;
        }
#pragma unroll
        for (int k = 0; k < HH; k += 32) {
            v16h av = load_tile_a(&hj[m0 * HJ_LD], HJ_LD, lane, k);   // LDS
            v16h b0 = load_tile_b(Wi, HH, lane, k);
            ai = __builtin_amdgcn_wmma_f32_16x16x32_f16(false, av, false, b0, (short)0, ai, false, false);
            v16h b1 = load_tile_b(Wf, HH, lane, k);
            af = __builtin_amdgcn_wmma_f32_16x16x32_f16(false, av, false, b1, (short)0, af, false, false);
            v16h b2 = load_tile_b(Wg, HH, lane, k);
            ag = __builtin_amdgcn_wmma_f32_16x16x32_f16(false, av, false, b2, (short)0, ag, false, false);
            v16h b3 = load_tile_b(Wo, HH, lane, k);
            ao = __builtin_amdgcn_wmma_f32_16x16x32_f16(false, av, false, b3, (short)0, ao, false, false);
        }
        __syncthreads();      // all hj reads complete before writes

        // register-resident cell update (PyTorch gate order i,f,g,o)
#pragma unroll
        for (int r = 0; r < 8; ++r) {
            const int m = m0 + mb + r;
            const float i_ = sigm(ai[r]);
            const float f_ = sigm(af[r]);
            const float g_ = tanhf(ag[r]);
            const float o_ = sigm(ao[r]);
            const float c  = f_ * cst[m * HH + jcol] + i_ * g_;
            const float h  = o_ * tanhf(c);
            cst[m * HH + jcol] = c;
            hj[m * HJ_LD + jcol] = (_Float16)h;
            if (hs16) hs16[((size_t)t * BB + m) * HH + jcol] = (_Float16)h;
            if (t == steps - 1) {
                if (h_out) h_out[m * HH + jcol] = h;
                if (c_out) c_out[m * HH + jcol] = c;
            }
        }
        __syncthreads();
    }
}

// ---------------------------------------------------------------------------
// FC projection: out[b,t,v] = hs16[row,:] . fcW16[v,:] + fc_b[v]
// row = (t-1)*32 + b, t = 1..63. Each wave: 2 m-tiles x 4 n-tiles macro-tile
// (8 accumulators; B reused across m, A across n). K=256 fully unrolled.
// ---------------------------------------------------------------------------
__global__ __launch_bounds__(256) void k_fc(const _Float16* __restrict__ A,    // [2016,256]
                                            const _Float16* __restrict__ W,    // [32000,256]
                                            const float* __restrict__ bias,    // [32000]
                                            float* __restrict__ out,           // [32,64,32000]
                                            int Ngroups, int totalJobs) {
    const int wave = threadIdx.x >> 5;
    const int lane = threadIdx.x & 31;
    const int job  = blockIdx.x * 8 + wave;
    if (job >= totalJobs) return;                  // wave-uniform exit
    const int m0 = (job / Ngroups) * 32;           // 2 m-tiles of 16
    const int n0 = (job % Ngroups) * 64;           // 4 n-tiles of 16
    v8f acc[8];
#pragma unroll
    for (int j = 0; j < 4; ++j) {
        const float bn = bias[n0 + 16 * j + (lane & 15)];
#pragma unroll
        for (int r = 0; r < 8; ++r) { acc[j][r] = bn; acc[4 + j][r] = bn; }
    }
#pragma unroll
    for (int k = 0; k < HH; k += 32) {
        v16h a0 = load_tile_a(A + (size_t)m0 * HH,        HH, lane, k);
        v16h a1 = load_tile_a(A + (size_t)(m0 + 16) * HH, HH, lane, k);
#pragma unroll
        for (int j = 0; j < 4; ++j) {
            v16h bv = load_tile_b(W + (size_t)(n0 + 16 * j) * HH, HH, lane, k);
            acc[j]     = __builtin_amdgcn_wmma_f32_16x16x32_f16(false, a0, false, bv,
                                                                (short)0, acc[j], false, false);
            acc[4 + j] = __builtin_amdgcn_wmma_f32_16x16x32_f16(false, a1, false, bv,
                                                                (short)0, acc[4 + j], false, false);
        }
    }
    const int mb = (lane >> 4) * 8;
#pragma unroll
    for (int j = 0; j < 8; ++j) {
        const int n  = n0 + 16 * (j & 3) + (lane & 15);
        const int mm = m0 + 16 * (j >> 2) + mb;
#pragma unroll
        for (int r = 0; r < 8; ++r) {
            const int row = mm + r;                // = (t-1)*32 + b
            const int b   = row & 31;
            const int t   = (row >> 5) + 1;
            out[((size_t)b * TT + t) * VV + n] = acc[j][r];
        }
    }
}

// Zero the t=0 output plane (harness poisons d_out).
__global__ void k_zero_t0(float* __restrict__ out) {
    const int i = blockIdx.x * blockDim.x + threadIdx.x;   // over 32*32000
    const int b = i / VV, v = i - b * VV;
    out[(size_t)b * TT * VV + v] = 0.0f;
}

// ---------------------------------------------------------------------------
// Host side
// ---------------------------------------------------------------------------
extern "C" void kernel_launch(void* const* d_in, const int* in_sizes, int n_in,
                              void* d_out, int out_size, void* d_ws, size_t ws_size,
                              hipStream_t stream) {
    (void)in_sizes; (void)n_in; (void)out_size; (void)ws_size;

    const int*   src     = (const int*)  d_in[0];
    const int*   trg     = (const int*)  d_in[1];
    const float* enc_emb = (const float*)d_in[2];
    const float* enc_Wih = (const float*)d_in[3];
    const float* enc_Whh = (const float*)d_in[4];
    const float* enc_b   = (const float*)d_in[5];
    const float* dec_emb = (const float*)d_in[6];
    const float* dec_Wih = (const float*)d_in[7];
    const float* dec_Whh = (const float*)d_in[8];
    const float* dec_b   = (const float*)d_in[9];
    const float* fc_W    = (const float*)d_in[10];
    const float* fc_b    = (const float*)d_in[11];
    float* out = (float*)d_out;

    // workspace carve-up (256B aligned)
    char* ws = (char*)d_ws;
    size_t off = 0;
    auto carve = [&](size_t bytes) -> char* {
        char* p = ws + off;
        off = (off + bytes + 255) & ~(size_t)255;
        return p;
    };
    _Float16* encWih16 = (_Float16*)carve((size_t)G4H * EE * 2);
    _Float16* encWhh16 = (_Float16*)carve((size_t)G4H * HH * 2);
    _Float16* decWih16 = (_Float16*)carve((size_t)G4H * EE * 2);
    _Float16* decWhh16 = (_Float16*)carve((size_t)G4H * HH * 2);
    _Float16* fcW16    = (_Float16*)carve((size_t)VV * HH * 2);
    _Float16* encX16   = (_Float16*)carve((size_t)SS * BB * EE * 2);
    _Float16* decX16   = (_Float16*)carve((size_t)(TT - 1) * BB * EE * 2);
    float*    encXG    = (float*)   carve((size_t)SS * BB * G4H * 4);
    float*    decXG    = (float*)   carve((size_t)(TT - 1) * BB * G4H * 4);
    float*    hbuf     = (float*)   carve((size_t)BB * HH * 4);
    float*    cbuf     = (float*)   carve((size_t)BB * HH * 4);
    _Float16* hs16     = (_Float16*)carve((size_t)(TT - 1) * BB * HH * 2);

    // 1) weight conversions f32 -> f16
    auto cvt = [&](const float* in, _Float16* o, int n) {
        k_f32_to_f16<<<(n + 255) / 256, 256, 0, stream>>>(in, o, n);
    };
    cvt(enc_Wih, encWih16, G4H * EE);
    cvt(enc_Whh, encWhh16, G4H * HH);
    cvt(dec_Wih, decWih16, G4H * EE);
    cvt(dec_Whh, decWhh16, G4H * HH);
    cvt(fc_W,    fcW16,    VV * HH);

    // 2) embedding gathers (encoder: all S steps; decoder: trg[:, :-1])
    k_gather_embed<<<SS * BB,       EE, 0, stream>>>(enc_emb, src, encX16, SS);
    k_gather_embed<<<(TT - 1) * BB, EE, 0, stream>>>(dec_emb, trg, decX16, TT);

    // 3) hoisted input GEMMs: x @ Wih^T  (M x 1024, K=128)
    {
        const int Mp = SS * BB / 32, Ng = G4H / 64;           // 64 x 16
        k_gemm_xw<<<(Mp * Ng + 7) / 8, 256, 0, stream>>>(encX16, encWih16, encXG, Mp, Ng);
    }
    {
        const int Mp = (TT - 1) * BB / 32, Ng = G4H / 64;     // 63 x 16
        k_gemm_xw<<<(Mp * Ng + 7) / 8, 256, 0, stream>>>(decX16, decWih16, decXG, Mp, Ng);
    }

    // 4) encoder recurrence (h0=c0=0), keep final h,c
    k_lstm<<<1, 1024, 0, stream>>>(encXG, encWhh16, enc_b,
                                   nullptr, nullptr, hbuf, cbuf, nullptr, SS);

    // 5) decoder recurrence seeded from encoder, store all hidden states (f16)
    k_lstm<<<1, 1024, 0, stream>>>(decXG, decWhh16, dec_b,
                                   hbuf, cbuf, nullptr, nullptr, hs16, TT - 1);

    // 6) t=0 plane zeros, then big FC projection into [B,T,V]
    k_zero_t0<<<(BB * VV) / 256, 256, 0, stream>>>(out);
    {
        const int Ng = VV / 64;                                // 500 groups of 4 n-tiles
        const int total = ((TT - 1) * BB / 32) * Ng;           // 63*500 = 31500
        k_fc<<<(total + 7) / 8, 256, 0, stream>>>(hs16, fcW16, fc_b, out, Ng, total);
    }
}
